// CrossAttentionSpanClassifier_1030792151080
// MI455X (gfx1250) — matrix operand order repaired and checked by reading, not compiled, not run on hardware
//
#include <hip/hip_runtime.h>
#include <hip/hip_bf16.h>
#include <math.h>

#define HID 768
#define BATCH 16
#define SEQ 512
#define NH 4
#define HD 192
#define NL 9

typedef __attribute__((ext_vector_type(16))) _Float16 v16h;
typedef __attribute__((ext_vector_type(8)))  float    v8f;

union F16Frag { v16h v; uint4 u[2]; };

__device__ __forceinline__ v16h make_frag(const _Float16* p0, const _Float16* p1) {
  F16Frag f;
  f.u[0] = *(const uint4*)p0;
  f.u[1] = *(const uint4*)p1;
  return f.v;
}

__device__ __forceinline__ v8f wmma_f16(v16h a, v16h b, v8f c) {
  // D = A(16x32 f16) * B(32x16 f16) + C(16x16 f32)
  return __builtin_amdgcn_wmma_f32_16x16x32_f16(false, a, false, b, (short)0, c,
                                                false, false);
}

// ---------------------------------------------------------------------------
// Kernel 1: Q/K/V = x @ W{q,k,v} + b   (f32 in, f16 out)
//   Q,K stored [bh][s][d] row-major; V stored transposed [bh][d][s]
//   128x64 block tile, each wave owns a 2x2 grid of 16x16 WMMA tiles.
// grid = (64 row-blocks, 12 col-blocks, 3 mats), block = 256 (8 waves)
// ---------------------------------------------------------------------------
__global__ __launch_bounds__(256) void qkv_kernel(
    const float* __restrict__ x,
    const float* __restrict__ Wq, const float* __restrict__ bq,
    const float* __restrict__ Wk, const float* __restrict__ bk,
    const float* __restrict__ Wv, const float* __restrict__ bv,
    _Float16* __restrict__ Qh, _Float16* __restrict__ Kh,
    _Float16* __restrict__ Vt)
{
  __shared__ __align__(16) _Float16 As[128][40];  // 128 rows x 32 k
  __shared__ __align__(16) _Float16 Bt[64][40];   // 64 n    x 32 k (W transposed)

  const int rowblk = blockIdx.x;
  const int colblk = blockIdx.y;
  const int mat    = blockIdx.z;
  const float* W    = (mat == 0) ? Wq : (mat == 1) ? Wk : Wv;
  const float* bias = (mat == 0) ? bq : (mat == 1) ? bk : bv;

  const int t = threadIdx.x;
  const int wave = t >> 5, lane = t & 31;
  const int M = lane & 15, hi = lane >> 4;
  const int mi0 = (wave & 3) * 2;    // m-tile pair: 0,2,4,6
  const int nj0 = (wave >> 2) * 2;   // n-tile pair: 0,2
  const int row0 = rowblk * 128;
  const int n0   = colblk * 64;

  v8f c[2][2];
#pragma unroll
  for (int im = 0; im < 2; ++im)
#pragma unroll
    for (int in = 0; in < 2; ++in) c[im][in] = (v8f){0,0,0,0,0,0,0,0};

  for (int k0 = 0; k0 < HID; k0 += 32) {
    {   // stage A tile (x, fp32 -> f16): 128x32, 16 elems/thread
      int r = t >> 1, cb = (t & 1) * 16;
      const float* src = x + (size_t)(row0 + r) * HID + k0 + cb;
#pragma unroll
      for (int q = 0; q < 16; ++q) As[r][cb + q] = (_Float16)src[q];
    }
    {   // stage B tile transposed: Bt[n][kk] = W[k0+kk][n0+n]
      int n = t & 63;
      int kk0 = (t >> 6) * 8;
#pragma unroll
      for (int q = 0; q < 8; ++q)
        Bt[n][kk0 + q] = (_Float16)W[(size_t)(k0 + kk0 + q) * HID + n0 + n];
    }
    __syncthreads();
    v16h a0 = make_frag(&As[(mi0 + 0) * 16 + M][hi * 16],
                        &As[(mi0 + 0) * 16 + M][hi * 16 + 8]);
    v16h a1 = make_frag(&As[(mi0 + 1) * 16 + M][hi * 16],
                        &As[(mi0 + 1) * 16 + M][hi * 16 + 8]);
    v16h b0 = make_frag(&Bt[(nj0 + 0) * 16 + M][hi * 16],
                        &Bt[(nj0 + 0) * 16 + M][hi * 16 + 8]);
    v16h b1 = make_frag(&Bt[(nj0 + 1) * 16 + M][hi * 16],
                        &Bt[(nj0 + 1) * 16 + M][hi * 16 + 8]);
    c[0][0] = wmma_f16(a0, b0, c[0][0]);
    c[0][1] = wmma_f16(a0, b1, c[0][1]);
    c[1][0] = wmma_f16(a1, b0, c[1][0]);
    c[1][1] = wmma_f16(a1, b1, c[1][1]);
    __syncthreads();
  }

  // epilogue: branch-free scatter. head/b are scalar per block.
  const int head  = colblk / 3;           // 192 = 3*64: col-block inside 1 head
  const int dbase = n0 - head * HD;
  const int b_    = row0 >> 9;            // 128 | 512 -> uniform per block
  const int s0    = row0 & 511;
  _Float16* dst;
  int sStr, dStr;
  if (mat == 0)      { dst = Qh; sStr = HD; dStr = 1; }
  else if (mat == 1) { dst = Kh; sStr = HD; dStr = 1; }
  else               { dst = Vt; sStr = 1;  dStr = SEQ; }
  dst += (size_t)(b_ * NH + head) * SEQ * HD;

#pragma unroll
  for (int im = 0; im < 2; ++im)
#pragma unroll
    for (int in = 0; in < 2; ++in) {
      int nloc = (nj0 + in) * 16 + (lane & 15);
      int d = dbase + nloc;
      float bv_ = bias[n0 + nloc];
#pragma unroll
      for (int j = 0; j < 8; ++j) {
        int s = s0 + (mi0 + im) * 16 + j + 8 * hi;
        dst[(size_t)s * sStr + (size_t)d * dStr] =
            (_Float16)(c[im][in][j] + bv_);
      }
    }
}

// ---------------------------------------------------------------------------
// Kernel 2: attention for one (bh, 32-row q-block)
//   scores = (QK^T + rel_bias)/sqrt(D) + dist_mask ; softmax ; ctx = P @ V
//   LDS = exactly 64KB score tile, reused in-place as f16 probabilities.
// grid = (16, 64), block = 256 (8 waves)
// ---------------------------------------------------------------------------
__global__ __launch_bounds__(256) void attn_kernel(
    const _Float16* __restrict__ Qh, const _Float16* __restrict__ Kh,
    const _Float16* __restrict__ Vt, _Float16* __restrict__ Ctx)
{
  __shared__ __align__(16) float Sc[32][512];   // 64 KB

  const int q0 = blockIdx.x * 32;
  const int bh = blockIdx.y;
  const int t = threadIdx.x, wave = t >> 5, lane = t & 31;
  const int N = lane & 15, hi = lane >> 4;

  const _Float16* Qb = Qh + (size_t)bh * SEQ * HD;
  const _Float16* Kb = Kh + (size_t)bh * SEQ * HD;
  const _Float16* Vb = Vt + (size_t)bh * HD * SEQ;

  const int mi = wave & 1;            // q 16-row tile: 0..1
  const int njbase = (wave >> 1) * 8; // 8 key tiles per wave
  const float invSqrtD = 0.07216878364870323f;  // 1/sqrt(192)

  // ---- scores: 32x512 via WMMA, K-dim = 192 ----
  v8f acc[8];
#pragma unroll
  for (int i = 0; i < 8; ++i) acc[i] = (v8f){0,0,0,0,0,0,0,0};

  for (int d0 = 0; d0 < HD; d0 += 32) {
    const _Float16* ap = Qb + (size_t)(q0 + mi * 16 + N) * HD + d0;
    v16h a = make_frag(ap + hi * 16, ap + hi * 16 + 8);
#pragma unroll
    for (int i = 0; i < 8; ++i) {
      const _Float16* bp =
          Kb + (size_t)((njbase + i) * 16 + N) * HD + d0 + hi * 16;
      v16h b = make_frag(bp, bp + 8);
      acc[i] = wmma_f16(a, b, acc[i]);
    }
  }
  // store with relative-position bias and distance mask
#pragma unroll
  for (int i = 0; i < 8; ++i) {
    int col = (njbase + i) * 16 + N;
#pragma unroll
    for (int j = 0; j < 8; ++j) {
      int m = mi * 16 + j + 8 * hi;
      float dist = fabsf((float)(q0 + m) - (float)col);
      float rel = __expf(-0.1f * fminf(dist, 5.0f));
      Sc[m][col] = (acc[i][j] + rel) * invSqrtD - 0.1f * dist;
    }
  }
  __syncthreads();

  // ---- softmax: 8 lanes per row ----
  const int r = t >> 3;
  const int sub = t & 7;
  float mx = -1e30f;
  for (int c = sub; c < 512; c += 8) mx = fmaxf(mx, Sc[r][c]);
#pragma unroll
  for (int m_ = 1; m_ < 8; m_ <<= 1) mx = fmaxf(mx, __shfl_xor(mx, m_, 32));
  float sum = 0.f;
  for (int c = sub; c < 512; c += 8) sum += __expf(Sc[r][c] - mx);
#pragma unroll
  for (int m_ = 1; m_ < 8; m_ <<= 1) sum += __shfl_xor(sum, m_, 32);
  float inv = 1.0f / sum;

  // ---- in-place f32 -> f16 probabilities (barrier-chunked) ----
  _Float16* ScH = (_Float16*)&Sc[0][0];   // half row stride = 1024
  for (int c0 = 0; c0 < 512; c0 += 128) {
    float vals[16];
    int cb = c0 + sub * 16;
#pragma unroll
    for (int q = 0; q < 16; ++q) vals[q] = __expf(Sc[r][cb + q] - mx) * inv;
    __syncthreads();
    uint* dst = (uint*)(ScH + (size_t)r * 1024 + cb);
#pragma unroll
    for (int q = 0; q < 8; ++q) {
      union { _Float16 h[2]; uint u; } pk;
      pk.h[0] = (_Float16)vals[2 * q];
      pk.h[1] = (_Float16)vals[2 * q + 1];
      dst[q] = pk.u;
    }
    __syncthreads();
  }

  // ---- ctx = P(32x512 f16, in LDS) @ V(512x192, transposed in ws) ----
  const int nj3 = (wave >> 1) * 3;   // 3 d-tiles per wave (12 total)
  v8f cc[3];
#pragma unroll
  for (int i = 0; i < 3; ++i) cc[i] = (v8f){0,0,0,0,0,0,0,0};

  for (int k0 = 0; k0 < SEQ; k0 += 32) {
    const _Float16* ap = ScH + (size_t)(mi * 16 + N) * 1024 + k0;
    v16h a = make_frag(ap + hi * 16, ap + hi * 16 + 8);
#pragma unroll
    for (int i = 0; i < 3; ++i) {
      const _Float16* bp =
          Vb + (size_t)((nj3 + i) * 16 + N) * SEQ + k0 + hi * 16;
      v16h b = make_frag(bp, bp + 8);
      cc[i] = wmma_f16(a, b, cc[i]);
    }
  }
  // store ctx as [b*S+s][HID] row-major f16 (A-matrix for Wo GEMM)
  const int b_ = bh >> 2, h_ = bh & 3;
#pragma unroll
  for (int i = 0; i < 3; ++i) {
    int dcol = (nj3 + i) * 16 + N;
#pragma unroll
    for (int j = 0; j < 8; ++j) {
      int s = q0 + mi * 16 + j + 8 * hi;
      Ctx[((size_t)(b_ * SEQ + s)) * HID + h_ * HD + dcol] = (_Float16)cc[i][j];
    }
  }
}

// ---------------------------------------------------------------------------
// Kernel 3a: Hpre = x + (ctx @ Wo + bo)
//   128x64 block tile, 2x2 WMMA tiles per wave; A read f16 direct from ws.
// grid = (64, 12), block = 256
// ---------------------------------------------------------------------------
__global__ __launch_bounds__(256) void proj_kernel(
    const _Float16* __restrict__ Ctx, const float* __restrict__ Wo,
    const float* __restrict__ bo, const float* __restrict__ x,
    float* __restrict__ Hpre)
{
  __shared__ __align__(16) _Float16 Bt[64][40];
  const int row0 = blockIdx.x * 128, n0 = blockIdx.y * 64;
  const int t = threadIdx.x, wave = t >> 5, lane = t & 31;
  const int M = lane & 15, hi = lane >> 4;
  const int mi0 = (wave & 3) * 2, nj0 = (wave >> 2) * 2;

  v8f c[2][2];
#pragma unroll
  for (int im = 0; im < 2; ++im)
#pragma unroll
    for (int in = 0; in < 2; ++in) c[im][in] = (v8f){0,0,0,0,0,0,0,0};

  for (int k0 = 0; k0 < HID; k0 += 32) {
    {
      int n = t & 63;
      int kk0 = (t >> 6) * 8;
#pragma unroll
      for (int q = 0; q < 8; ++q)
        Bt[n][kk0 + q] = (_Float16)Wo[(size_t)(k0 + kk0 + q) * HID + n0 + n];
    }
    __syncthreads();
    const _Float16* ap0 = Ctx + (size_t)(row0 + (mi0 + 0) * 16 + M) * HID + k0;
    const _Float16* ap1 = Ctx + (size_t)(row0 + (mi0 + 1) * 16 + M) * HID + k0;
    v16h a0 = make_frag(ap0 + hi * 16, ap0 + hi * 16 + 8);
    v16h a1 = make_frag(ap1 + hi * 16, ap1 + hi * 16 + 8);
    v16h b0 = make_frag(&Bt[(nj0 + 0) * 16 + M][hi * 16],
                        &Bt[(nj0 + 0) * 16 + M][hi * 16 + 8]);
    v16h b1 = make_frag(&Bt[(nj0 + 1) * 16 + M][hi * 16],
                        &Bt[(nj0 + 1) * 16 + M][hi * 16 + 8]);
    c[0][0] = wmma_f16(a0, b0, c[0][0]);
    c[0][1] = wmma_f16(a0, b1, c[0][1]);
    c[1][0] = wmma_f16(a1, b0, c[1][0]);
    c[1][1] = wmma_f16(a1, b1, c[1][1]);
    __syncthreads();
  }
#pragma unroll
  for (int im = 0; im < 2; ++im)
#pragma unroll
    for (int in = 0; in < 2; ++in) {
      int ng = n0 + (nj0 + in) * 16 + (lane & 15);
      float bv_ = bo[ng];
#pragma unroll
      for (int j = 0; j < 8; ++j) {
        int rowg = row0 + (mi0 + im) * 16 + j + 8 * hi;
        Hpre[(size_t)rowg * HID + ng] =
            c[im][in][j] + bv_ + x[(size_t)rowg * HID + ng];
      }
    }
}

// ---------------------------------------------------------------------------
// Kernel 3b: LayerNorm + span logits. One wave per row. grid=1024, block=256
// ---------------------------------------------------------------------------
__global__ __launch_bounds__(256) void ln_logits_kernel(
    const float* __restrict__ Hpre, const float* __restrict__ ln_g,
    const float* __restrict__ ln_b, const float* __restrict__ Ws,
    const float* __restrict__ bs, float* __restrict__ span_ws,
    float* __restrict__ out)
{
  const int t = threadIdx.x, wave = t >> 5, lane = t & 31;
  const int row = blockIdx.x * 8 + wave;
  const float* hr = Hpre + (size_t)row * HID;

  float v[24];
  float s = 0.f;
#pragma unroll
  for (int q = 0; q < 24; ++q) { v[q] = hr[lane + 32 * q]; s += v[q]; }
#pragma unroll
  for (int m_ = 1; m_ < 32; m_ <<= 1) s += __shfl_xor(s, m_, 32);
  float mu = s * (1.0f / 768.0f);
  float var = 0.f;
#pragma unroll
  for (int q = 0; q < 24; ++q) { float d = v[q] - mu; var += d * d; }
#pragma unroll
  for (int m_ = 1; m_ < 32; m_ <<= 1) var += __shfl_xor(var, m_, 32);
  float rstd = rsqrtf(var * (1.0f / 768.0f) + 1e-5f);

  float acc[NL];
#pragma unroll
  for (int l = 0; l < NL; ++l) acc[l] = 0.f;
#pragma unroll
  for (int q = 0; q < 24; ++q) {
    int k = lane + 32 * q;
    float hn = (v[q] - mu) * rstd * ln_g[k] + ln_b[k];
#pragma unroll
    for (int l = 0; l < NL; ++l) acc[l] += hn * Ws[k * NL + l];
  }
#pragma unroll
  for (int l = 0; l < NL; ++l)
#pragma unroll
    for (int m_ = 1; m_ < 32; m_ <<= 1) acc[l] += __shfl_xor(acc[l], m_, 32);

  if (lane == 0) {
#pragma unroll
    for (int l = 0; l < NL; ++l) {
      float lg = acc[l] + bs[l];
      span_ws[(size_t)row * NL + l] = lg;   // pristine copy for argmax
      out[(size_t)row * NL + l] = lg;
    }
  }
}

// ---------------------------------------------------------------------------
// Kernel 4: entity-bias bump from argmax of previous row's pristine logits
// ---------------------------------------------------------------------------
__global__ void bump_kernel(const float* __restrict__ span_ws,
                            const float* __restrict__ entity_bias,
                            float* __restrict__ out)
{
  int idx = blockIdx.x * blockDim.x + threadIdx.x;
  int total = BATCH * (SEQ - 1);
  if (idx >= total) return;
  int b = idx / (SEQ - 1);
  int j = idx % (SEQ - 1) + 1;
  const float* prev = span_ws + ((size_t)b * SEQ + j - 1) * NL;
  int best = 0;
  float bv_ = prev[0];
#pragma unroll
  for (int l = 1; l < NL; ++l) {
    float xv = prev[l];
    if (xv > bv_) { bv_ = xv; best = l; }   // first-max = jnp.argmax semantics
  }
  if (best == 1)   // B_PERSON
    out[((size_t)b * SEQ + j) * NL + 2] += 2.0f * entity_bias[2];  // I_PERSON
}

// ---------------------------------------------------------------------------
extern "C" void kernel_launch(void* const* d_in, const int* in_sizes, int n_in,
                              void* d_out, int out_size, void* d_ws,
                              size_t ws_size, hipStream_t stream) {
  (void)in_sizes; (void)n_in; (void)out_size; (void)ws_size;
  const float* x    = (const float*)d_in[0];
  const float* Wq   = (const float*)d_in[1];
  const float* bq   = (const float*)d_in[2];
  const float* Wk   = (const float*)d_in[3];
  const float* bk   = (const float*)d_in[4];
  const float* Wv   = (const float*)d_in[5];
  const float* bv   = (const float*)d_in[6];
  const float* Wo   = (const float*)d_in[7];
  const float* bo   = (const float*)d_in[8];
  const float* ln_g = (const float*)d_in[9];
  const float* ln_b = (const float*)d_in[10];
  const float* Ws   = (const float*)d_in[11];
  const float* bs   = (const float*)d_in[12];
  const float* eb   = (const float*)d_in[13];
  float* out = (float*)d_out;

  char* ws = (char*)d_ws;
  size_t off = 0;
  auto alloc = [&](size_t bytes) -> void* {
    void* p = ws + off;
    off += (bytes + 255) & ~(size_t)255;
    return p;
  };
  const size_t qkvHalves = (size_t)BATCH * NH * SEQ * HD;   // 6,291,456
  _Float16* Qh   = (_Float16*)alloc(qkvHalves * 2);
  _Float16* Kh   = (_Float16*)alloc(qkvHalves * 2);
  _Float16* Vt   = (_Float16*)alloc(qkvHalves * 2);
  _Float16* Ctx  = (_Float16*)alloc((size_t)BATCH * SEQ * HID * 2);
  float*    Hpre = (float*)alloc((size_t)BATCH * SEQ * HID * 4);
  float*    Span = (float*)alloc((size_t)BATCH * SEQ * NL * 4);

  qkv_kernel<<<dim3(64, 12, 3), 256, 0, stream>>>(x, Wq, bq, Wk, bk, Wv, bv,
                                                  Qh, Kh, Vt);
  attn_kernel<<<dim3(16, 64), 256, 0, stream>>>(Qh, Kh, Vt, Ctx);
  proj_kernel<<<dim3(64, 12), 256, 0, stream>>>(Ctx, Wo, bo, x, Hpre);
  ln_logits_kernel<<<1024, 256, 0, stream>>>(Hpre, ln_g, ln_b, Ws, bs, Span, out);
  bump_kernel<<<(BATCH * (SEQ - 1) + 255) / 256, 256, 0, stream>>>(Span, eb, out);
}
